// ConvAttention_81879256531068
// MI455X (gfx1250) — matrix operand order repaired and checked by reading, not compiled, not run on hardware
//
#include <hip/hip_runtime.h>

// ConvAttention on MI455X (gfx1250): flash-attention, f16 WMMA with f32 accum.
// N = 8192 tokens, D = 128. Scores never leave registers; K/V stay L2-resident.
// Row-sums via WMMA against a ones-matrix; wave-shared running max with lazy
// rescale; all 8 V B-fragments prefetched across the softmax dependency chain.

typedef _Float16 v16h __attribute__((ext_vector_type(16)));
typedef _Float16 v8h  __attribute__((ext_vector_type(8)));
typedef float    v8f  __attribute__((ext_vector_type(8)));

#define N_TOK 8192
#define DH    128
#define QK_SCALE 0.08838834764831845f   // 1/sqrt(128)

#define SHUF16(lo, hi) __builtin_shufflevector((lo), (hi), 0,1,2,3,4,5,6,7,8,9,10,11,12,13,14,15)

// ---------------------------------------------------------------------------
// Prep 1: V -> Vt[128][8192] f16 (channel-major == x's native layout, coalesced)
__global__ __launch_bounds__(256) void prep_vt(const float* __restrict__ x,
                                               _Float16* __restrict__ vt) {
    int idx = blockIdx.x * 256 + threadIdx.x;   // [0, 128*8192)
    int c = idx >> 13;
    int n = idx & (N_TOK - 1);
    int b = n >> 10, hw = n & 1023;
    vt[idx] = (_Float16)x[((b * 384 + c) << 10) + hw];
}

// ---------------------------------------------------------------------------
// Prep 2: K,Q -> token-major [8192][128] f16 via LDS tile transpose.
// blockIdx.x: bit0 = (0:K ch 128..255, 1:Q ch 256..383), rest = 32-token tile.
__global__ __launch_bounds__(256) void prep_qk(const float* __restrict__ x,
                                               _Float16* __restrict__ kh,
                                               _Float16* __restrict__ qh) {
    __shared__ _Float16 tile[32 * 130];
    int isQ = blockIdx.x & 1;
    int n0  = (blockIdx.x >> 1) * 32;           // 32 tokens, same batch (32|1024)
    int b = n0 >> 10, hw0 = n0 & 1023;
    int chbase = isQ ? 256 : 128;
    for (int i = threadIdx.x; i < 32 * 128; i += 256) {
        int c = i >> 5, t = i & 31;             // coalesced read along hw
        float v = x[((b * 384 + chbase + c) << 10) + hw0 + t];
        if (isQ) v *= QK_SCALE;
        tile[t * 130 + c] = (_Float16)v;
    }
    __syncthreads();
    _Float16* dst = isQ ? qh : kh;
    for (int i = threadIdx.x; i < 32 * 128; i += 256) {
        int t = i >> 7, c = i & 127;            // coalesced 256B row writes
        dst[(n0 + t) * DH + c] = tile[t * 130 + c];
    }
}

// ---------------------------------------------------------------------------
// Flash attention: one wave (32 threads) per 16-row query block, 512 blocks.
__global__ __launch_bounds__(32) void flash_attn(const _Float16* __restrict__ qh,
                                                 const _Float16* __restrict__ kh,
                                                 const _Float16* __restrict__ vt,
                                                 float* __restrict__ out) {
    __shared__ __align__(16) _Float16 psc[16 * 40];   // padded P bounce tile

    const int lane = threadIdx.x & 31;
    const int half = lane >> 4;          // 0 | 1
    const int l15  = lane & 15;
    const int q0   = blockIdx.x * 16;

    // --- Q A-fragments (16x128 as 4 chunks of K=32), contiguous k order, same as the B frags:
    // lane l: row m = l&15, elems 0..15 -> k = 16*(l>>4)+0..15.
    v16h qa[4];
    {
        const _Float16* qrow = qh + (q0 + l15) * DH + half * 16;
        #pragma unroll
        for (int c = 0; c < 4; ++c) {
            v8h lo = *(const v8h*)(qrow + c * 32);
            v8h hi = *(const v8h*)(qrow + c * 32 + 8);
            qa[c] = SHUF16(lo, hi);
        }
    }

    // Ones B-fragment: P(16x32) x ones(32x16) -> row-sums replicated over cols,
    // landing in exactly the C/D layout of the L accumulator.
    v16h ones;
    #pragma unroll
    for (int i = 0; i < 16; ++i) ones[i] = (_Float16)1.0f;

    const v8f vzero = {0.f, 0.f, 0.f, 0.f, 0.f, 0.f, 0.f, 0.f};
    v8f o[8];
    #pragma unroll
    for (int s = 0; s < 8; ++s) o[s] = vzero;
    v8f  L = vzero;            // per-row softmax denominators (D layout)
    float M = -1e30f;          // wave-shared running max bound

    const int kbB = half * 16;           // B-frag: lane l, k = 16*(l>>4)+e

    for (int kt = 0; kt < N_TOK / 32; ++kt) {
        const int k0 = kt * 32;

        // --- S = Q . K^T : two 16x16 tiles (keys k0..15, k0+16..31)
        v8f s0 = vzero, s1 = vzero;
        #pragma unroll
        for (int c = 0; c < 4; ++c) {
            const _Float16* kr0 = kh + (k0 + l15) * DH + c * 32 + kbB;
            v8h a0 = *(const v8h*)kr0;
            v8h a1 = *(const v8h*)(kr0 + 8);
            v16h b0 = SHUF16(a0, a1);
            s0 = __builtin_amdgcn_wmma_f32_16x16x32_f16(false, qa[c], false, b0,
                                                        (short)0, s0, false, false);
            const _Float16* kr1 = kh + (k0 + 16 + l15) * DH + c * 32 + kbB;
            v8h a2 = *(const v8h*)kr1;
            v8h a3 = *(const v8h*)(kr1 + 8);
            v16h b1 = SHUF16(a2, a3);
            s1 = __builtin_amdgcn_wmma_f32_16x16x32_f16(false, qa[c], false, b1,
                                                        (short)0, s1, false, false);
        }

        // --- prefetch ALL 8 V B-fragments now: addresses depend only on k0,
        // so these 16 b128 loads fly during the whole softmax chain below.
        v16h vb[8];
        #pragma unroll
        for (int s = 0; s < 8; ++s) {
            const _Float16* vr = vt + (s * 16 + l15) * N_TOK + k0 + kbB;
            v8h lo = *(const v8h*)vr;
            v8h hi = *(const v8h*)(vr + 8);
            vb[s] = SHUF16(lo, hi);
        }

        // --- wave-shared tile max: per-lane tree over 16 values, then a
        // 5-round butterfly. Any shared upper bound keeps exp() stable;
        // softmax itself is shift-invariant so the result is exact.
        float tm = fmaxf(s0[0], s1[0]);
        #pragma unroll
        for (int r = 1; r < 8; ++r) tm = fmaxf(tm, fmaxf(s0[r], s1[r]));
        #pragma unroll
        for (int off = 1; off < 32; off <<= 1)
            tm = fmaxf(tm, __shfl_xor(tm, off, 32));

        if (tm > M) {                     // uniformly-taken, rare after warmup
            float rs = __expf(M - tm);    // == 0 on the first tile
            M = tm;
            L *= rs;
            #pragma unroll
            for (int s = 0; s < 8; ++s) o[s] *= rs;
        }

        // --- P = exp(S - M)
        #pragma unroll
        for (int r = 0; r < 8; ++r) {
            s0[r] = __expf(s0[r] - M);
            s1[r] = __expf(s1[r] - M);
        }

        // --- P: C/D layout -> A layout via per-wave LDS bounce (in-order DS).
        #pragma unroll
        for (int r = 0; r < 8; ++r) {
            int m = r + half * 8;
            psc[m * 40 + l15]      = (_Float16)s0[r];
            psc[m * 40 + 16 + l15] = (_Float16)s1[r];
        }
        __builtin_amdgcn_wave_barrier();
        v16h pa;
        {
            const _Float16* pr = psc + l15 * 40 + half * 16;   // 16B aligned
            v8h lo = *(const v8h*)pr;
            v8h hi = *(const v8h*)(pr + 8);
            pa = SHUF16(lo, hi);
        }
        __builtin_amdgcn_wave_barrier();

        // --- L += rowsum(P) on the matrix core
        L = __builtin_amdgcn_wmma_f32_16x16x32_f16(false, pa, false, ones,
                                                   (short)0, L, false, false);

        // --- O += P . V : fragments already resident, WMMAs run back-to-back
        #pragma unroll
        for (int s = 0; s < 8; ++s) {
            o[s] = __builtin_amdgcn_wmma_f32_16x16x32_f16(false, pa, false, vb[s],
                                                          (short)0, o[s], false, false);
        }
    }

    // --- normalize and scatter to channel-major output [B,128,H,W]
    #pragma unroll
    for (int s = 0; s < 8; ++s) {
        #pragma unroll
        for (int r = 0; r < 8; ++r) {
            int m  = r + half * 8;
            int n  = q0 + m;
            int b  = n >> 10, hw = n & 1023;
            int ch = s * 16 + l15;
            out[((b * 128 + ch) << 10) + hw] = o[s][r] / L[r];
        }
    }
}

// ---------------------------------------------------------------------------
extern "C" void kernel_launch(void* const* d_in, const int* in_sizes, int n_in,
                              void* d_out, int out_size, void* d_ws, size_t ws_size,
                              hipStream_t stream) {
    const float* x = (const float*)d_in[0];
    float* out = (float*)d_out;

    _Float16* qh = (_Float16*)d_ws;              // [8192][128] token-major (scaled)
    _Float16* kh = qh + (size_t)N_TOK * DH;      // [8192][128] token-major
    _Float16* vt = kh + (size_t)N_TOK * DH;      // [128][8192] channel-major

    prep_vt<<<(128 * N_TOK) / 256, 256, 0, stream>>>(x, vt);
    prep_qk<<<(N_TOK / 32) * 2, 256, 0, stream>>>(x, kh, qh);
    flash_attn<<<N_TOK / 16, 32, 0, stream>>>(qh, kh, vt, out);
}
